// StochaPolicy_50379966382218
// MI455X (gfx1250) — matrix operand order repaired and checked by reading, not compiled, not run on hardware
//
#include <hip/hip_runtime.h>

#define NB     32768
#define DIM    256
#define KC     1024
#define AOUT   32
#define TM     16
#define AS     (DIM + 8)     // obs-tile LDS stride (bf16 elems)
#define PS     (KC + 8)      // phi LDS stride (bf16 elems)
#define THREADS 256

typedef __attribute__((ext_vector_type(16))) __bf16 v16bf;
typedef __attribute__((ext_vector_type(8)))  float  v8f;
typedef __attribute__((ext_vector_type(4)))  unsigned int u32x4;
typedef __attribute__((ext_vector_type(4)))  float  f32x4;

union BF16x16 { v16bf v; u32x4 q[2]; };
struct BGroup { BF16x16 h[4]; BF16x16 l[4]; };   // 4 column-tiles, hi+lo (64 VGPRs)

__device__ __forceinline__ unsigned short f32_to_bf16(float f) {
  unsigned int u = __float_as_uint(f);
  u += 0x7fffu + ((u >> 16) & 1u);      // round-to-nearest-even
  return (unsigned short)(u >> 16);
}
__device__ __forceinline__ float bf16_to_f32(unsigned short h) {
  return __uint_as_float(((unsigned int)h) << 16);
}

// ---------------------------------------------------------------------------
// Prep: convert C (both paths) to bf16 hi/lo, w to bf16, and compute c2[k].
// ---------------------------------------------------------------------------
__global__ __launch_bounds__(THREADS)
void prep_kernel(const float* __restrict__ C0, const float* __restrict__ C1,
                 const float* __restrict__ W0, const float* __restrict__ W1,
                 float* __restrict__ c2,            // [2*KC]
                 unsigned short* __restrict__ Chi,  // [2*KC*DIM]
                 unsigned short* __restrict__ Clo,  // [2*KC*DIM]
                 unsigned short* __restrict__ Whi)  // [2*AOUT*KC]
{
  __shared__ float ssum[THREADS];
  const int r = blockIdx.x;            // 0..2047
  const int t = threadIdx.x;           // 0..255 == DIM
  const float* src = (r < KC) ? (C0 + (size_t)r * DIM) : (C1 + (size_t)(r - KC) * DIM);
  float v = src[t];
  unsigned short hi = f32_to_bf16(v);
  Chi[(size_t)r * DIM + t] = hi;
  Clo[(size_t)r * DIM + t] = f32_to_bf16(v - bf16_to_f32(hi));
  ssum[t] = v * v;
  __syncthreads();
  for (int s = THREADS / 2; s > 0; s >>= 1) {
    if (t < s) ssum[t] += ssum[t + s];
    __syncthreads();
  }
  if (t == 0) c2[r] = ssum[0];

  if (r < 256) {                        // 256*256 == 2*AOUT*KC elements
    int idx = r * 256 + t;
    float wv = (idx < AOUT * KC) ? W0[idx] : W1[idx - AOUT * KC];
    Whi[idx] = f32_to_bf16(wv);
  }
}

// ---------------------------------------------------------------------------
// Fragment helpers
// ---------------------------------------------------------------------------
__device__ __forceinline__ void load_bgroup(BGroup& G,
                                            const unsigned short* __restrict__ Cph,
                                            const unsigned short* __restrict__ Cpl,
                                            int nbase, int dofs) {
#pragma unroll
  for (int t = 0; t < 4; ++t) {
    const unsigned short* bh = Cph + (size_t)(nbase + t * 16) * DIM + dofs;
    const unsigned short* bl = Cpl + (size_t)(nbase + t * 16) * DIM + dofs;
    G.h[t].q[0] = *(const u32x4*)(bh);
    G.h[t].q[1] = *(const u32x4*)(bh + 8);
    G.l[t].q[0] = *(const u32x4*)(bl);
    G.l[t].q[1] = *(const u32x4*)(bl + 8);
  }
}

// bf16x3: terms ordered so consecutive WMMAs hit different accumulators.
__device__ __forceinline__ void wmma_group(v8f* acc, const BGroup& G,
                                           const BF16x16& Ah, const BF16x16& Al) {
#pragma unroll
  for (int t = 0; t < 4; ++t)
    acc[t] = __builtin_amdgcn_wmma_f32_16x16x32_bf16(false, Ah.v, false, G.h[t].v,
                                                     (short)0, acc[t], false, false);
#pragma unroll
  for (int t = 0; t < 4; ++t)
    acc[t] = __builtin_amdgcn_wmma_f32_16x16x32_bf16(false, Al.v, false, G.h[t].v,
                                                     (short)0, acc[t], false, false);
#pragma unroll
  for (int t = 0; t < 4; ++t)
    acc[t] = __builtin_amdgcn_wmma_f32_16x16x32_bf16(false, Ah.v, false, G.l[t].v,
                                                     (short)0, acc[t], false, false);
}

// ---------------------------------------------------------------------------
// Main fused kernel: 16 rows per block, 8 waves; each wave owns 128 centroid
// columns processed as two double-buffered groups of 4 WMMA tiles.
// ---------------------------------------------------------------------------
__global__ __launch_bounds__(THREADS)
void rbf_main(const float* __restrict__ obs,
              const float* __restrict__ sig0, const float* __restrict__ bias0,
              const float* __restrict__ sig1, const float* __restrict__ bias1,
              const float* __restrict__ c2ws,
              const unsigned short* __restrict__ Chi,
              const unsigned short* __restrict__ Clo,
              const unsigned short* __restrict__ Whi,
              float* __restrict__ out)
{
  __shared__ float x2lds[TM];
  __shared__ float outlds[TM * AOUT];
  __shared__ unsigned short ahi_l[TM * AS];
  __shared__ unsigned short alo_l[TM * AS];
  __shared__ unsigned short phil[TM * PS];

  const int tid  = threadIdx.x;
  const int lane = tid & 31;
  const int wave = tid >> 5;           // 0..7
  const int row0 = blockIdx.x * TM;
  const int l15  = lane & 15;
  const bool lo16 = (lane < 16);

  if (tid < TM) x2lds[tid] = 0.f;
  __syncthreads();

  // --- convert this block's 16x256 obs tile to bf16 hi/lo, accumulate x2 ---
  {
    const int m  = tid >> 4;           // 0..15
    const int d0 = (tid & 15) * 16;    // 16 consecutive elements
    const f32x4* p = (const f32x4*)(obs + (size_t)(row0 + m) * DIM + d0);
    float s = 0.f;
    #pragma unroll
    for (int q = 0; q < 4; ++q) {
      f32x4 f = p[q];
      #pragma unroll
      for (int e = 0; e < 4; ++e) {
        float v = f[e];
        s += v * v;
        unsigned short h = f32_to_bf16(v);
        ahi_l[m * AS + d0 + q * 4 + e] = h;
        alo_l[m * AS + d0 + q * 4 + e] = f32_to_bf16(v - bf16_to_f32(h));
      }
    }
    atomicAdd(&x2lds[m], s);           // ds_add_f32
  }
  __syncthreads();

  float x2v[8];
  {
    const int mb = lo16 ? 0 : 8;       // C/D tile rows this lane holds
    #pragma unroll
    for (int j = 0; j < 8; ++j) x2v[j] = x2lds[mb + j];
  }

  const int ncol0 = wave * 128;        // this wave's 128 centroid columns
  const int koffA = lo16 ? 0 : 16;     // A-fragment k-offset (contiguous k = 16h + s)
  const int koffB = lo16 ? 0 : 16;     // B-fragment k-offset (contiguous k = 16h + s)
  const int nfA   = ncol0 + l15;       // group 0 first column for this lane
  const int nfB   = ncol0 + 64 + l15;  // group 1 first column
  const v8f vzero = {0.f, 0.f, 0.f, 0.f, 0.f, 0.f, 0.f, 0.f};

  for (int path = 0; path < 2; ++path) {
    const float* sig  = path ? sig1 : sig0;
    const float* bias = path ? bias1 : bias0;
    const float* c2   = c2ws + path * KC;
    const unsigned short* Cph = Chi + (size_t)path * KC * DIM;
    const unsigned short* Cpl = Clo + (size_t)path * KC * DIM;
    const unsigned short* Wp  = Whi + (size_t)path * AOUT * KC;

    outlds[tid] = 0.f;
    outlds[tid + THREADS] = 0.f;
    __syncthreads();

    // ---------------- Stage 1: P = x @ C^T (bf16x3, software pipelined) ----
    v8f acc[8];
    #pragma unroll
    for (int t = 0; t < 8; ++t) acc[t] = vzero;

    BGroup B0, B1;
    load_bgroup(B0, Cph, Cpl, nfA, 0 + koffB);   // prime the pipeline

    #pragma unroll
    for (int ks = 0; ks < 8; ++ks) {
      const int d0 = ks * 32;
      // A fragments for this k-step (LDS, short latency)
      BF16x16 Ah, Al;
      const unsigned short* ap  = &ahi_l[l15 * AS + d0 + koffA];
      const unsigned short* alp = &alo_l[l15 * AS + d0 + koffA];
      Ah.q[0] = *(const u32x4*)(ap);       Ah.q[1] = *(const u32x4*)(ap + 8);
      Al.q[0] = *(const u32x4*)(alp);      Al.q[1] = *(const u32x4*)(alp + 8);

      // prefetch group 1 of this k-step, then compute group 0
      load_bgroup(B1, Cph, Cpl, nfB, d0 + koffB);
      wmma_group(acc, B0, Ah, Al);

      // prefetch group 0 of the next k-step, then compute group 1
      if (ks < 7) load_bgroup(B0, Cph, Cpl, nfA, d0 + 32 + koffB);
      wmma_group(acc + 4, B1, Ah, Al);
    }

    // ---------------- phi = exp(-r/(2|sigma|)) -> LDS (bf16) ---------------
    {
      const int mb = lo16 ? 0 : 8;
      #pragma unroll
      for (int t = 0; t < 8; ++t) {
        const int n = ncol0 + ((t & 1) ? 64 : 0) + (t >> 1) * 16 + l15;
        const v8f& a = acc[(t & 1) ? 4 + (t >> 1) : (t >> 1)];
        const float c2n = c2[n];
        const float sc  = 0.5f / fabsf(sig[n]);
        #pragma unroll
        for (int j = 0; j < 8; ++j) {
          float r  = x2v[j] - 2.f * a[j] + c2n;
          float ph = __expf(-r * sc);
          phil[(mb + j) * PS + n] = f32_to_bf16(ph);
        }
      }
    }
    __syncthreads();

    // ---------------- Stage 2: out = phi @ w^T (K split across waves) ------
    v8f oacc[2];
    oacc[0] = vzero; oacc[1] = vzero;
    #pragma unroll
    for (int s = 0; s < 4; ++s) {
      const int k0 = ncol0 + s * 32;
      BF16x16 Pa, Wb0, Wb1;
      const unsigned short* wp0 = Wp + (size_t)(0 * 16 + l15) * KC + k0 + koffB;
      const unsigned short* wp1 = Wp + (size_t)(1 * 16 + l15) * KC + k0 + koffB;
      Wb0.q[0] = *(const u32x4*)(wp0);  Wb0.q[1] = *(const u32x4*)(wp0 + 8);
      Wb1.q[0] = *(const u32x4*)(wp1);  Wb1.q[1] = *(const u32x4*)(wp1 + 8);
      const unsigned short* pp = &phil[l15 * PS + k0 + koffA];
      Pa.q[0] = *(const u32x4*)(pp);
      Pa.q[1] = *(const u32x4*)(pp + 8);
      oacc[0] = __builtin_amdgcn_wmma_f32_16x16x32_bf16(false, Pa.v, false, Wb0.v,
                                                        (short)0, oacc[0], false, false);
      oacc[1] = __builtin_amdgcn_wmma_f32_16x16x32_bf16(false, Pa.v, false, Wb1.v,
                                                        (short)0, oacc[1], false, false);
    }
    {
      const int mb = lo16 ? 0 : 8;
      #pragma unroll
      for (int t = 0; t < 2; ++t) {
        const int a = t * 16 + l15;
        #pragma unroll
        for (int j = 0; j < 8; ++j)
          atomicAdd(&outlds[(mb + j) * AOUT + a], oacc[t][j]);   // ds_add_f32
      }
    }
    __syncthreads();

    // ---------------- finalize: bias (+ clip/exp for std path) -------------
    for (int e = tid; e < TM * AOUT; e += THREADS) {
      const int m = e >> 5, a = e & 31;
      float v = outlds[e] + bias[a];
      if (path == 1) v = __expf(fminf(fmaxf(v, -20.f), 2.f));
      out[(size_t)(row0 + m) * (2 * AOUT) + path * AOUT + a] = v;
    }
    __syncthreads();                    // protect LDS reuse for next path
  }
}

// ---------------------------------------------------------------------------
extern "C" void kernel_launch(void* const* d_in, const int* in_sizes, int n_in,
                              void* d_out, int out_size, void* d_ws, size_t ws_size,
                              hipStream_t stream) {
  const float* obs  = (const float*)d_in[0];
  const float* mC   = (const float*)d_in[1];
  const float* msig = (const float*)d_in[2];
  const float* mw   = (const float*)d_in[3];
  const float* mb   = (const float*)d_in[4];
  const float* sC   = (const float*)d_in[5];
  const float* ssig = (const float*)d_in[6];
  const float* sw   = (const float*)d_in[7];
  const float* sb   = (const float*)d_in[8];
  float* out = (float*)d_out;

  // workspace layout (total ~2.2 MB)
  char* ws = (char*)d_ws;
  float*          c2  = (float*)ws;                                  //   8 KB
  unsigned short* Chi = (unsigned short*)(ws + 8192);                //   1 MB
  unsigned short* Clo = (unsigned short*)(ws + 8192 + 1048576);      //   1 MB
  unsigned short* Whi = (unsigned short*)(ws + 8192 + 2097152);      // 128 KB

  prep_kernel<<<2 * KC, THREADS, 0, stream>>>(mC, sC, mw, sw, c2, Chi, Clo, Whi);
  rbf_main<<<NB / TM, THREADS, 0, stream>>>(obs, msig, mb, ssig, sb,
                                            c2, Chi, Clo, Whi, out);
}
